// NextDiT_22660247454474
// MI455X (gfx1250) — matrix operand order repaired and checked by reading, not compiled, not run on hardware
//
#include <hip/hip_runtime.h>
#include <hip/hip_bf16.h>
#include <math.h>

typedef __bf16 bf16_t;
typedef __attribute__((ext_vector_type(16))) __bf16 v16bf;
typedef __attribute__((ext_vector_type(8)))  float v8f;
typedef __attribute__((ext_vector_type(8)))  unsigned int v8u;

#define B_  2
#define S_  2048
#define D_  2048
#define H_  16
#define HD_ 128
#define YL_ 256
#define YD_ 2048
#define LN_EPS 1e-5f

// ---------------- WMMA fragment helpers (gfx1250 wave32 layouts) ----------------
// 16-bit A matrix 16x32: lanes 0-15 = rows M, K in {0..7,16..23}; lanes 16-31 same rows, K in {8..15,24..31}.
// Caller passes pointer already offset by (half*8) in K; pairs are K-contiguous dwords.
static __device__ inline v16bf load_a_frag(const bf16_t* p) {
  const unsigned int* q = (const unsigned int*)p;
  v8u u;
  u[0] = q[0]; u[1] = q[1]; u[2] = q[2]; u[3] = q[3];   // K +0..7
  u[4] = q[8]; u[5] = q[9]; u[6] = q[10]; u[7] = q[11]; // K +16..23
  return __builtin_bit_cast(v16bf, u);
}
// 16-bit B matrix 32x16: lane%16 = column N; lanes 0-15 K in {0..7,16..23}, lanes 16-31 K in {8..15,24..31} (pointer offset by half*8).
static __device__ inline v16bf load_b_frag(const bf16_t* p) {
  const unsigned int* q = (const unsigned int*)p;
  v8u u;
#pragma unroll
  for (int i = 0; i < 8; ++i) u[i] = q[(i & 3) + (i >> 2) * 8];
  return __builtin_bit_cast(v16bf, u);
}
static __device__ inline v8f wmma_bf16(v16bf a, v16bf b, v8f c) {
  return __builtin_amdgcn_wmma_f32_16x16x32_bf16(false, a, false, b, (short)0, c, false, false);
}
static __device__ inline float hmax16(float v) {
  v = fmaxf(v, __shfl_xor(v, 1, 32));
  v = fmaxf(v, __shfl_xor(v, 2, 32));
  v = fmaxf(v, __shfl_xor(v, 4, 32));
  v = fmaxf(v, __shfl_xor(v, 8, 32));
  return v;
}
static __device__ inline float hsum16(float v) {
  v += __shfl_xor(v, 1, 32);
  v += __shfl_xor(v, 2, 32);
  v += __shfl_xor(v, 4, 32);
  v += __shfl_xor(v, 8, 32);
  return v;
}

// ---------------- TDM L2 warm: one tensor_load_to_lds per workgroup ----------------
// Pulls a (rowsPerTile x rowUnits) tile of 8-byte elements through the cache
// hierarchy into LDS. The LDS contents are intentionally unconsumed; this is a
// correctness-neutral prefetch of the weight matrices into the 192MB L2 before
// the GEMM chain. Descriptor packed per CDNA5 D# spec (group0/group1).
typedef __attribute__((ext_vector_type(4))) unsigned int u32x4_t;
typedef __attribute__((ext_vector_type(4))) int          i32x4_t;
typedef __attribute__((ext_vector_type(8))) int          i32x8_t;

__global__ void tdm_warm(const void* __restrict__ base, unsigned rowUnits /*8B units per row*/,
                         unsigned rowsPerTile, unsigned totalRows, float* __restrict__ sink) {
  __shared__ unsigned char stage[32 * 1024];
  const unsigned row0 = blockIdx.x * rowsPerTile;
  unsigned long long gaddr = (unsigned long long)base + (unsigned long long)row0 * rowUnits * 8ull;

  u32x4_t g0;
  g0[0] = 1u;                                            // count=1, user descriptor
  g0[1] = 0u;                                            // lds_addr = 0 (start of stage)
  g0[2] = (unsigned)(gaddr & 0xFFFFFFFFu);               // global_addr[31:0]
  g0[3] = ((unsigned)(gaddr >> 32) & 0x01FFFFFFu) | 0x80000000u; // global_addr[56:32] | type=2

  i32x8_t g1;
  g1[0] = (int)(3u << 16);                               // workgroup_mask=0, data_size=8B
  g1[1] = (int)((rowUnits & 0xFFFFu) << 16);             // tensor_dim0[15:0] @ bits 63:48
  g1[2] = (int)(((rowUnits >> 16) & 0xFFFFu) |
                ((totalRows & 0xFFFFu) << 16));          // tensor_dim0[31:16] | tensor_dim1[15:0]
  g1[3] = (int)(((totalRows >> 16) & 0xFFFFu) |
                ((rowUnits & 0xFFFFu) << 16));           // tensor_dim1[31:16] | tile_dim0
  g1[4] = (int)(rowsPerTile & 0xFFFFu);                  // tile_dim1 (tile_dim2=0)
  g1[5] = (int)rowUnits;                                 // tensor_dim0_stride[31:0]
  g1[6] = (int)((rowUnits & 0xFFFFu) << 16);             // stride0[47:32]=0 | stride1[15:0]
  g1[7] = 0;                                             // stride1[47:16]

  i32x4_t z4 = {0, 0, 0, 0};
#if __clang_major__ >= 23
  i32x8_t z8 = {0, 0, 0, 0, 0, 0, 0, 0};
  __builtin_amdgcn_tensor_load_to_lds(g0, g1, z4, z4, z8, 0);
#else
  __builtin_amdgcn_tensor_load_to_lds(g0, g1, z4, z4, 0);
#endif
  __builtin_amdgcn_s_wait_tensorcnt(0);
  __syncthreads();
  // keep the LDS allocation live; condition is practically never true and the
  // sink target is overwritten by later GEMMs anyway.
  if (threadIdx.x == 0 && ((volatile unsigned*)stage)[0] == 0xDEADBEEFu) sink[0] += 1.f;
}

// ---------------- elementwise converts / transposes ----------------
__global__ void convert_bf16(const float* __restrict__ in, bf16_t* __restrict__ out, size_t n) {
  size_t i = (size_t)blockIdx.x * blockDim.x + threadIdx.x;
  if (i < n) out[i] = (bf16_t)in[i];
}

// w: (K,N) f32  ->  wT: (N,K) bf16   (so GEMM B-fragments read contiguous K)
__global__ void transpose_w_bf16(const float* __restrict__ w, bf16_t* __restrict__ wT, int K, int N) {
  size_t i = (size_t)blockIdx.x * blockDim.x + threadIdx.x;
  if (i < (size_t)K * N) {
    int n = (int)(i / K), k = (int)(i % K);
    wT[i] = (bf16_t)w[(size_t)k * N + n];
  }
}

// v: (B,Skv,H,HD) f32 -> vT: (B,H,HD,Skv) bf16  (PV B-fragments read contiguous keys)
__global__ void transpose_v_bf16(const float* __restrict__ v, bf16_t* __restrict__ vT, int Skv, size_t n) {
  size_t i = (size_t)blockIdx.x * blockDim.x + threadIdx.x;
  if (i >= n) return;
  int s = (int)(i % Skv); size_t rest = i / Skv;
  int d = (int)(rest % HD_); rest /= HD_;
  int h = (int)(rest % H_);  int b = (int)(rest / H_);
  vT[i] = (bf16_t)v[(((size_t)b * Skv + s) * H_ + h) * HD_ + d];
}

// ---------------- bf16 WMMA GEMM: C(MxN) f32 = A(MxK) bf16 * WT(NxK) bf16 ----------------
// one wave per 32x64 C tile: 2 A-fragments x 4 B-fragments -> 8 WMMAs per K-chunk,
// with every B-fragment reused across both M-halves.
__global__ void gemm_bf16(const bf16_t* __restrict__ A, const bf16_t* __restrict__ WT,
                          float* __restrict__ C, int M, int N, int K) {
  const int lane = threadIdx.x & 31;
  const int half = lane >> 4, lid = lane & 15;
  const int n0 = blockIdx.x * 64;
  const int m0 = blockIdx.y * 32;
  (void)M;
  v8f acc[2][4];
#pragma unroll
  for (int m = 0; m < 2; ++m)
#pragma unroll
    for (int t = 0; t < 4; ++t)
#pragma unroll
      for (int r = 0; r < 8; ++r) acc[m][t][r] = 0.f;

  const bf16_t* arow0 = A + (size_t)(m0 + lid) * K + half * 8;
  const bf16_t* arow1 = arow0 + (size_t)16 * K;
  for (int k0 = 0; k0 < K; k0 += 32) {
    __builtin_prefetch(arow0 + k0 + 128, 0, 1);
    __builtin_prefetch(arow1 + k0 + 128, 0, 1);
    v16bf af0 = load_a_frag(arow0 + k0);
    v16bf af1 = load_a_frag(arow1 + k0);
#pragma unroll
    for (int t = 0; t < 4; ++t) {
      const bf16_t* brow = WT + (size_t)(n0 + t * 16 + lid) * K + k0 + half * 8;
      v16bf bf = load_b_frag(brow);
      acc[0][t] = wmma_bf16(af0, bf, acc[0][t]);
      acc[1][t] = wmma_bf16(af1, bf, acc[1][t]);
    }
  }
#pragma unroll
  for (int m = 0; m < 2; ++m)
#pragma unroll
    for (int t = 0; t < 4; ++t)
#pragma unroll
      for (int r = 0; r < 8; ++r)
        C[(size_t)(m0 + m * 16 + r + half * 8) * N + n0 + t * 16 + lid] = acc[m][t][r];
}

// ---------------- fused LayerNorm (+optional RoPE) -> bf16 ----------------
__global__ void ln_rope_bf16(const float* __restrict__ x, const float* __restrict__ w,
                             const float* __restrict__ bias,
                             const float* __restrict__ cosT, const float* __restrict__ sinT,
                             bf16_t* __restrict__ out, int seqlen, int do_rope) {
  const int row = blockIdx.x;          // over B*seqlen
  const int s   = row % seqlen;
  const int tid = threadIdx.x;         // 256 threads
  __shared__ float red[256];
  const float* xr = x + (size_t)row * D_;

  float sum = 0.f;
  for (int i = tid; i < D_; i += 256) sum += xr[i];
  red[tid] = sum; __syncthreads();
  for (int off = 128; off > 0; off >>= 1) { if (tid < off) red[tid] += red[tid + off]; __syncthreads(); }
  const float mean = red[0] / (float)D_;
  __syncthreads();

  float vs = 0.f;
  for (int i = tid; i < D_; i += 256) { float d = xr[i] - mean; vs += d * d; }
  red[tid] = vs; __syncthreads();
  for (int off = 128; off > 0; off >>= 1) { if (tid < off) red[tid] += red[tid + off]; __syncthreads(); }
  const float rstd = rsqrtf(red[0] / (float)D_ + LN_EPS);

  bf16_t* orow = out + (size_t)row * D_;
  for (int p = tid; p < D_ / 2; p += 256) {
    int e0 = 2 * p, e1 = 2 * p + 1;
    float a = (xr[e0] - mean) * rstd * w[e0] + bias[e0];
    float c = (xr[e1] - mean) * rstd * w[e1] + bias[e1];
    if (do_rope) {
      int i  = p & (HD_ / 2 - 1);
      float co = cosT[(size_t)s * (HD_ / 2) + i];
      float si = sinT[(size_t)s * (HD_ / 2) + i];
      float na = a * co - c * si;
      float nc = a * si + c * co;
      a = na; c = nc;
    }
    orow[e0] = (bf16_t)a;
    orow[e1] = (bf16_t)c;
  }
}

// ---------------- flash attention: one wave per 16-row q tile ----------------
// q,k: (B,*,H,HD) bf16 ; vT: (B,H,HD,Skv) bf16 ; out: (B,S,H,HD) f32
__global__ void attn_wmma(const bf16_t* __restrict__ q, const bf16_t* __restrict__ k,
                          const bf16_t* __restrict__ vT, const unsigned char* __restrict__ mask,
                          const float* __restrict__ gate, float* __restrict__ out,
                          int S, int Skv, int accumulate) {
  const int lane = threadIdx.x & 31;
  const int half = lane >> 4, lid = lane & 15;
  const int q0 = blockIdx.x * 16;
  const int h  = blockIdx.y;
  const int b  = blockIdx.z;
  const float scale = 0.08838834764831845f;   // 1/sqrt(128)

  // preload 16xHD q tile as 4 A-fragments (K chunks of 32)
  v16bf qf[4];
  const bf16_t* qrow = q + ((size_t)((size_t)b * S + q0 + lid) * D_) + h * HD_ + half * 8;
#pragma unroll
  for (int kc = 0; kc < 4; ++kc) qf[kc] = load_a_frag(qrow + kc * 32);

  v8f oacc[8];
#pragma unroll
  for (int dt = 0; dt < 8; ++dt)
#pragma unroll
    for (int r = 0; r < 8; ++r) oacc[dt][r] = 0.f;
  float mrow[8], lrow[8];
#pragma unroll
  for (int r = 0; r < 8; ++r) { mrow[r] = -3.0e38f; lrow[r] = 0.f; }

  __shared__ bf16_t pbuf[16 * 32];

  for (int kb = 0; kb < Skv; kb += 32) {
    // ---- scores: two 16x16 tiles over 32 keys, K = HD reduced by 4 WMMAs each
    v8f sc[2];
#pragma unroll
    for (int t = 0; t < 2; ++t)
#pragma unroll
      for (int r = 0; r < 8; ++r) sc[t][r] = 0.f;
#pragma unroll
    for (int t = 0; t < 2; ++t) {
      const bf16_t* krow = k + ((size_t)((size_t)b * Skv + kb + t * 16 + lid) * D_) + h * HD_ + half * 8;
#pragma unroll
      for (int kc = 0; kc < 4; ++kc) {
        v16bf bf = load_b_frag(krow + kc * 32);
        sc[t] = wmma_bf16(qf[kc], bf, sc[t]);
      }
    }
    // ---- mask + scale (column = key = kb + t*16 + lid)
#pragma unroll
    for (int t = 0; t < 2; ++t) {
      bool mok = mask[(size_t)b * Skv + kb + t * 16 + lid] != 0;
#pragma unroll
      for (int r = 0; r < 8; ++r) sc[t][r] = mok ? sc[t][r] * scale : -3.0e38f;
    }
    // ---- online softmax per row (M = r + half*8; reduce across 16 lanes of the half)
    float p0[8], p1[8];
#pragma unroll
    for (int r = 0; r < 8; ++r) {
      float mt   = hmax16(fmaxf(sc[0][r], sc[1][r]));
      float mnew = fmaxf(mrow[r], mt);
      float alpha = __expf(mrow[r] - mnew);
      mrow[r] = mnew;
      p0[r] = __expf(sc[0][r] - mnew);
      p1[r] = __expf(sc[1][r] - mnew);
      lrow[r] = lrow[r] * alpha + hsum16(p0[r] + p1[r]);
#pragma unroll
      for (int dt = 0; dt < 8; ++dt) oacc[dt][r] *= alpha;
    }
    // ---- re-layout P (C fragment -> A fragment) through LDS
    __syncthreads();
#pragma unroll
    for (int r = 0; r < 8; ++r) {
      pbuf[(r + half * 8) * 32 + lid]      = (bf16_t)p0[r];
      pbuf[(r + half * 8) * 32 + 16 + lid] = (bf16_t)p1[r];
    }
    __syncthreads();
    v8u pu;
    {
      const unsigned int* pl = (const unsigned int*)pbuf;
      int base = lid * 16 + half * 4;   // dwords: row lid, K base half*8
      pu[0] = pl[base + 0]; pu[1] = pl[base + 1]; pu[2] = pl[base + 2]; pu[3] = pl[base + 3];
      pu[4] = pl[base + 8]; pu[5] = pl[base + 9]; pu[6] = pl[base + 10]; pu[7] = pl[base + 11];
    }
    v16bf pa = __builtin_bit_cast(v16bf, pu);
    // ---- P @ V : 8 output d-tiles of 16
#pragma unroll
    for (int dt = 0; dt < 8; ++dt) {
      const bf16_t* vrow = vT + ((size_t)((size_t)b * H_ + h) * HD_ + dt * 16 + lid) * Skv + kb + half * 8;
      v16bf vf = load_b_frag(vrow);
      oacc[dt] = wmma_bf16(pa, vf, oacc[dt]);
    }
    __syncthreads();
  }

  float g = 1.0f;
  if (gate) g = tanhf(gate[h]);
#pragma unroll
  for (int dt = 0; dt < 8; ++dt)
#pragma unroll
    for (int r = 0; r < 8; ++r) {
      float val = oacc[dt][r] / lrow[r];
      size_t idx = ((size_t)((size_t)b * S + q0 + r + half * 8) * D_) + h * HD_ + dt * 16 + lid;
      if (accumulate) out[idx] += g * val; else out[idx] = val;
    }
}

// ---------------- host launcher ----------------
extern "C" void kernel_launch(void* const* d_in, const int* in_sizes, int n_in,
                              void* d_out, int out_size, void* d_ws, size_t ws_size,
                              hipStream_t stream) {
  (void)in_sizes; (void)n_in; (void)out_size; (void)ws_size;
  const float* x    = (const float*)d_in[0];
  const unsigned char* x_mask = (const unsigned char*)d_in[1];
  const float* y    = (const float*)d_in[2];
  const unsigned char* y_mask = (const unsigned char*)d_in[3];
  const float* fcos = (const float*)d_in[4];
  const float* fsin = (const float*)d_in[5];
  const float* wq   = (const float*)d_in[6];
  const float* wk   = (const float*)d_in[7];
  const float* wv   = (const float*)d_in[8];
  const float* wo   = (const float*)d_in[9];
  const float* wky  = (const float*)d_in[10];
  const float* wvy  = (const float*)d_in[11];
  const float* gate = (const float*)d_in[12];
  const float* qn_w = (const float*)d_in[13];
  const float* qn_b = (const float*)d_in[14];
  const float* kn_w = (const float*)d_in[15];
  const float* kn_b = (const float*)d_in[16];
  const float* kyn_w = (const float*)d_in[17];
  const float* kyn_b = (const float*)d_in[18];

  char* ws = (char*)d_ws;
  size_t off = 0;
  auto alloc = [&](size_t bytes) -> void* {
    void* p = ws + off;
    off += (bytes + 255) & ~(size_t)255;
    return p;
  };

  const size_t nx  = (size_t)B_ * S_ * D_;     // 8M elements
  const size_t ny  = (size_t)B_ * YL_ * YD_;   // 1M
  const size_t nyv = (size_t)B_ * YL_ * D_;

  bf16_t* xb    = (bf16_t*)alloc(nx * 2);
  bf16_t* yb    = (bf16_t*)alloc(ny * 2);
  bf16_t* wqT   = (bf16_t*)alloc((size_t)D_ * D_ * 2);
  bf16_t* wkT   = (bf16_t*)alloc((size_t)D_ * D_ * 2);
  bf16_t* wvT   = (bf16_t*)alloc((size_t)D_ * D_ * 2);
  bf16_t* woT   = (bf16_t*)alloc((size_t)D_ * D_ * 2);
  bf16_t* wkyT  = (bf16_t*)alloc((size_t)YD_ * D_ * 2);
  bf16_t* wvyT  = (bf16_t*)alloc((size_t)YD_ * D_ * 2);
  float*  tmp   = (float*)alloc(nx * 4);
  bf16_t* qb    = (bf16_t*)alloc(nx * 2);
  bf16_t* kbb   = (bf16_t*)alloc(nx * 2);
  bf16_t* vTb   = (bf16_t*)alloc(nx * 2);
  bf16_t* ykb   = (bf16_t*)alloc(nyv * 2);
  bf16_t* yvT   = (bf16_t*)alloc(nyv * 2);
  float*  attn  = (float*)alloc(nx * 4);
  bf16_t* attnb = (bf16_t*)alloc(nx * 2);

  const int T = 256;

  // TDM warm: stream all six fp32 weight matrices through L2 (32KB LDS tiles,
  // 4 rows x 1024 x 8B per workgroup; 2048 rows per weight -> 512 blocks each).
  tdm_warm<<<512, 32, 0, stream>>>(wq,  1024u, 4u, 2048u, tmp);
  tdm_warm<<<512, 32, 0, stream>>>(wk,  1024u, 4u, 2048u, tmp);
  tdm_warm<<<512, 32, 0, stream>>>(wv,  1024u, 4u, 2048u, tmp);
  tdm_warm<<<512, 32, 0, stream>>>(wo,  1024u, 4u, 2048u, tmp);
  tdm_warm<<<512, 32, 0, stream>>>(wky, 1024u, 4u, 2048u, tmp);
  tdm_warm<<<512, 32, 0, stream>>>(wvy, 1024u, 4u, 2048u, tmp);

  convert_bf16<<<dim3((unsigned)((nx + T - 1) / T)), T, 0, stream>>>(x, xb, nx);
  convert_bf16<<<dim3((unsigned)((ny + T - 1) / T)), T, 0, stream>>>(y, yb, ny);

  const size_t nw = (size_t)D_ * D_;
  dim3 gw((unsigned)((nw + T - 1) / T));
  transpose_w_bf16<<<gw, T, 0, stream>>>(wq,  wqT,  D_,  D_);
  transpose_w_bf16<<<gw, T, 0, stream>>>(wk,  wkT,  D_,  D_);
  transpose_w_bf16<<<gw, T, 0, stream>>>(wv,  wvT,  D_,  D_);
  transpose_w_bf16<<<gw, T, 0, stream>>>(wo,  woT,  D_,  D_);
  transpose_w_bf16<<<gw, T, 0, stream>>>(wky, wkyT, YD_, D_);
  transpose_w_bf16<<<gw, T, 0, stream>>>(wvy, wvyT, YD_, D_);

  dim3 gx(D_ / 64, (B_ * S_) / 32);
  dim3 gy(D_ / 64, (B_ * YL_) / 32);

  // Q = LN(x@wq) + RoPE
  gemm_bf16<<<gx, 32, 0, stream>>>(xb, wqT, tmp, B_ * S_, D_, D_);
  ln_rope_bf16<<<B_ * S_, 256, 0, stream>>>(tmp, qn_w, qn_b, fcos, fsin, qb, S_, 1);
  // K = LN(x@wk) + RoPE
  gemm_bf16<<<gx, 32, 0, stream>>>(xb, wkT, tmp, B_ * S_, D_, D_);
  ln_rope_bf16<<<B_ * S_, 256, 0, stream>>>(tmp, kn_w, kn_b, fcos, fsin, kbb, S_, 1);
  // V = x@wv, transposed for PV
  gemm_bf16<<<gx, 32, 0, stream>>>(xb, wvT, tmp, B_ * S_, D_, D_);
  transpose_v_bf16<<<dim3((unsigned)((nx + T - 1) / T)), T, 0, stream>>>(tmp, vTb, S_, nx);
  // yK = LN(y@wk_y), no RoPE
  gemm_bf16<<<gy, 32, 0, stream>>>(yb, wkyT, tmp, B_ * YL_, D_, YD_);
  ln_rope_bf16<<<B_ * YL_, 256, 0, stream>>>(tmp, kyn_w, kyn_b, nullptr, nullptr, ykb, YL_, 0);
  // yV = y@wv_y, transposed
  gemm_bf16<<<gy, 32, 0, stream>>>(yb, wvyT, tmp, B_ * YL_, D_, YD_);
  transpose_v_bf16<<<dim3((unsigned)((nyv + T - 1) / T)), T, 0, stream>>>(tmp, yvT, YL_, nyv);

  // self attention, then gated cross attention accumulated on top
  dim3 ga(S_ / 16, H_, B_);
  attn_wmma<<<ga, 32, 0, stream>>>(qb, kbb, vTb, x_mask, nullptr, attn, S_, S_, 0);
  attn_wmma<<<ga, 32, 0, stream>>>(qb, ykb, yvT, y_mask, gate, attn, S_, YL_, 1);

  // output projection
  convert_bf16<<<dim3((unsigned)((nx + T - 1) / T)), T, 0, stream>>>(attn, attnb, nx);
  gemm_bf16<<<gx, 32, 0, stream>>>(attnb, woT, (float*)d_out, B_ * S_, D_, D_);
}
